// QLSTMGen130_65481071397013
// MI455X (gfx1250) — matrix operand order repaired and checked by reading, not compiled, not run on hardware
//
#include <hip/hip_runtime.h>
#include <hip/hip_bf16.h>

// ---------------------------------------------------------------------------
// LSTM: T=1024, B=64, I=256, H=512.  Per step: pre = tanh(comb @ W + b),
// gates f,i,g,o -> c,h update.  comb = [x_t (256) | h_prev (512)] -> K = 768.
// Persistent-grid kernel: 64 WGs x 256 thr (8 waves) = 512 waves,
// one 16x16 WMMA tile per wave per step (4 row tiles x 128 col tiles).
// Activations use gfx1250's hardware V_TANH_F32 (TRANS32) when available.
// ---------------------------------------------------------------------------

#define LT   1024
#define LB   64
#define LI   256
#define LH   512
#define LK   (LI + LH)          // 768
#define NWG  64
#define NTHR 256
#define LDA  (LK + 8)           // LDS row pitch in bf16 (776) -> 1552B, 16B aligned

typedef __attribute__((ext_vector_type(16))) __bf16 v16bf;
typedef __attribute__((ext_vector_type(8)))  float  v8f;

union FragBF16 { uint4 u[2]; v16bf v; };

__device__ __forceinline__ unsigned short f2bf(float x) {
    unsigned u = __float_as_uint(x);
    unsigned r = u + 0x7FFFu + ((u >> 16) & 1u);   // round-to-nearest-even
    return (unsigned short)(r >> 16);
}

// ---- fast activations: single TRANS32 op on gfx1250 when builtin exists ----
__device__ __forceinline__ float tanh_fast(float x) {
#if __has_builtin(__builtin_amdgcn_tanhf)
    return __builtin_amdgcn_tanhf(x);            // v_tanh_f32
#elif __has_builtin(__builtin_amdgcn_tanh_f32)
    return __builtin_amdgcn_tanh_f32(x);         // v_tanh_f32
#else
    return tanhf(x);
#endif
}

__device__ __forceinline__ float sigmoid_fast(float x) {
#if __has_builtin(__builtin_amdgcn_tanhf) || __has_builtin(__builtin_amdgcn_tanh_f32)
    // sigma(x) = 0.5*tanh(x/2) + 0.5  -> 1 TRANS + 2 VALU
    return __builtin_fmaf(tanh_fast(0.5f * x), 0.5f, 0.5f);
#elif __has_builtin(__builtin_amdgcn_exp2f) && __has_builtin(__builtin_amdgcn_rcpf)
    float e = __builtin_amdgcn_exp2f(x * -1.44269504088896340736f);
    return __builtin_amdgcn_rcpf(1.0f + e);
#else
    return 1.0f / (1.0f + __expf(-x));
#endif
}

// -------------------------- prep kernels -----------------------------------

// zero h/c state + barrier words
__global__ void qlstm_init_state(float* hstate, float* cstate, unsigned* bar) {
    int i = blockIdx.x * blockDim.x + threadIdx.x;
    if (i < LB * LH) { hstate[i] = 0.0f; cstate[i] = 0.0f; }
    if (i == 0) { bar[0] = 0u; bar[1] = 0u; }
}

// W [4][768][512] f32  ->  Wt [4][512][768] bf16  (so B-fragments are
// contiguous 16-bf16 runs along K per output column)
__global__ void qlstm_conv_w(const float* __restrict__ W,
                             unsigned short* __restrict__ Wt) {
    int idx = blockIdx.x * blockDim.x + threadIdx.x;   // 4*768*512 threads
    int g = idx / (LK * LH);
    int r = idx - g * (LK * LH);
    int d = r / LH;
    int h = r - d * LH;
    Wt[((g * LH + h) * LK) + d] = f2bf(W[idx]);
}

// -------------------------- device-wide barrier -----------------------------

__device__ __forceinline__ void grid_barrier(unsigned* cnt, unsigned* gen) {
    __syncthreads();
    if (threadIdx.x == 0) {
        __threadfence();
        unsigned g = ((volatile unsigned*)gen)[0];
        if (atomicAdd(cnt, 1u) == (unsigned)(NWG - 1)) {
            *cnt = 0u;
            __threadfence();
            atomicAdd(gen, 1u);
        } else {
            while (((volatile unsigned*)gen)[0] == g) {
                __builtin_amdgcn_s_sleep(1);
            }
        }
        __threadfence();
    }
    __syncthreads();
}

// -------------------------- persistent LSTM kernel --------------------------

__global__ void __launch_bounds__(NTHR)
qlstm_persistent(const float* __restrict__ x,        // [T,B,I]
                 const float* __restrict__ bias,     // [4,H]
                 const unsigned short* __restrict__ Wt, // [4,H,K] bf16
                 float* __restrict__ hstate,         // [B,H]
                 float* __restrict__ cstate,         // [B,H]
                 float* __restrict__ preAct,         // [4,B,H]
                 unsigned* __restrict__ bar,
                 float* __restrict__ out) {          // [T,B,H] ++ [B,H] ++ [B,H]
    __shared__ __align__(16) unsigned short comb[16 * LDA];   // 16 rows x 776 bf16

    const int tid   = threadIdx.x;
    const int lane  = tid & 31;
    const int wave  = tid >> 5;              // 0..7
    const int wg    = blockIdx.x;            // 0..63
    const int btile = wg & 3;                // 16-row batch tile
    const int cgrp  = wg >> 2;               // 0..15
    const int col   = cgrp * 8 + wave;       // 0..127 column tile
    const int g     = col >> 5;              // gate 0..3
    const int hb    = (col & 31) << 4;       // h base within gate

    const int n     = lane & 15;             // output column within tile
    const int khalf = lane >> 4;             // 0: low K half, 1: high K half

    // per-lane B base: Wt[g][hb+n][*] — contiguous along K
    const unsigned short* brow = Wt + ((size_t)(g * LH + hb + n)) * LK;
    const float bv = bias[g * LH + hb + n];

    for (int t = 0; t < LT; ++t) {
        // ---- stage comb = [x_t rows | h_prev rows] into LDS as bf16 ----
        {
            const float* xrow = x + (size_t)t * (LB * LI);
            for (int idx = tid; idx < 16 * LK; idx += NTHR) {
                int r = idx / LK;
                int c = idx - r * LK;
                int b = btile * 16 + r;
                float v = (c < LI) ? xrow[b * LI + c]
                                   : hstate[b * LH + (c - LI)];
                comb[r * LDA + c] = f2bf(v);
            }
        }
        __syncthreads();

        // ---- 16x16 tile GEMM: K-loop of 24 bf16-WMMA (K=32 each) ----
        v8f acc = {};
        const unsigned short* crow = comb + (lane & 15) * LDA;
#pragma unroll 4
        for (int kc = 0; kc < LK / 32; ++kc) {
            const int k0 = kc * 32;
            // A fragment (16x32 bf16) from LDS, matching B's K order:
            // lanes 0-15: K {0-15}; lanes 16-31: K {16-31}
            FragBF16 a;
            a.u[0] = *(const uint4*)(crow + k0 + khalf * 16);
            a.u[1] = *(const uint4*)(crow + k0 + khalf * 16 + 8);
            // B fragment (32x16 bf16): lane n holds column hb+n,
            // K run [k0 + 16*khalf, +16) contiguous in Wt
            FragBF16 b;
            const unsigned short* bp = brow + k0 + khalf * 16;
            b.u[0] = *(const uint4*)(bp);
            b.u[1] = *(const uint4*)(bp + 8);
            // prefetch next K chunk of the streamed B matrix
            __builtin_prefetch((const void*)(bp + 32), 0, 3);
            acc = __builtin_amdgcn_wmma_f32_16x16x32_bf16(
                false, a.v, false, b.v, (short)0, acc, false, false);
        }

        // ---- pre-activation tanh + scatter to preAct[g][b][h] ----
        {
            const int mbase = khalf * 8;
            const int h = hb + n;
#pragma unroll
            for (int j = 0; j < 8; ++j) {
                int b = btile * 16 + mbase + j;
                preAct[((g * LB + b) * LH) + h] = tanh_fast(acc[j] + bv);
            }
        }

        grid_barrier(bar, bar + 1);

        // ---- elementwise gate update over all B*H elements ----
        {
            int gt = wg * NTHR + tid;                  // 0..16383
            for (int e = gt; e < LB * LH; e += NWG * NTHR) {
                float pf = preAct[(0 * LB * LH) + e];
                float pi = preAct[(1 * LB * LH) + e];
                float pg = preAct[(2 * LB * LH) + e];
                float po = preAct[(3 * LB * LH) + e];
                float f  = sigmoid_fast(pf);
                float i  = sigmoid_fast(pi);
                float gg = tanh_fast(pg);
                float o  = sigmoid_fast(po);
                float c  = __builtin_fmaf(f, cstate[e], i * gg);
                float hn = o * tanh_fast(c);
                cstate[e] = c;
                hstate[e] = hn;
                out[(size_t)t * (LB * LH) + e] = hn;
                if (t == LT - 1) {
                    out[(size_t)LT * (LB * LH) + e]            = hn;  // final hx
                    out[(size_t)LT * (LB * LH) + LB * LH + e]  = c;   // final cx
                }
            }
        }

        grid_barrier(bar, bar + 1);
    }
}

// -------------------------- launcher ----------------------------------------

extern "C" void kernel_launch(void* const* d_in, const int* in_sizes, int n_in,
                              void* d_out, int out_size, void* d_ws, size_t ws_size,
                              hipStream_t stream) {
    const float* x    = (const float*)d_in[0];   // [T,B,I]
    const float* W    = (const float*)d_in[1];   // [4,K,H]
    const float* bias = (const float*)d_in[2];   // [4,H]
    float* out = (float*)d_out;

    // workspace layout
    char* ws = (char*)d_ws;
    unsigned short* Wt = (unsigned short*)(ws);                 // 4*512*768*2 = 3,145,728
    float* hstate = (float*)(ws + 3145728);                     // 131,072
    float* cstate = (float*)(ws + 3145728 + 131072);            // 131,072
    float* preAct = (float*)(ws + 3145728 + 262144);            // 524,288
    unsigned* bar = (unsigned*)(ws + 3145728 + 262144 + 524288);

    (void)in_sizes; (void)n_in; (void)out_size; (void)ws_size;

    qlstm_init_state<<<(LB * LH + 255) / 256, 256, 0, stream>>>(hstate, cstate, bar);

    int wElems = 4 * LK * LH;                                   // 1,572,864
    qlstm_conv_w<<<wElems / 256, 256, 0, stream>>>(W, Wt);

    qlstm_persistent<<<NWG, NTHR, 0, stream>>>(x, bias, Wt, hstate, cstate,
                                               preAct, bar, out);
}
